// MMDInception_15796889715484
// MI455X (gfx1250) — compile-verified, matrix operand order checked by reading, not run on hardware
//
#include <hip/hip_runtime.h>
#include <hip/hip_bf16.h>

// ---------------- problem constants ----------------
#define NROWS 8192          // rows of Z = [real; gen]
#define HALFROWS 4096
#define KDIM 2048
#define BM 128
#define BN 128
#define BK 32
#define LDS_STRIDE 40       // halves per LDS tile row (32 data + 8 pad -> conflict-free b128)
#define TILE_HALVES (BM * LDS_STRIDE)
#define KSTEPS (KDIM / BK)  // 64

typedef __attribute__((ext_vector_type(16))) _Float16 v16h;
typedef __attribute__((ext_vector_type(8)))  _Float16 v8h;
typedef __attribute__((ext_vector_type(8)))  float    v8f;

// exact parameter type of the async builtin: int __attribute__((vector_size(16)))
typedef int v4i_ __attribute__((__vector_size__(16)));
typedef __attribute__((address_space(1))) v4i_ gv4i;
typedef __attribute__((address_space(3))) v4i_ lv4i;

// ---------------- CDNA5 async copy helpers ----------------
__device__ __forceinline__ void async_b128(void* lds, const void* g) {
#if __has_builtin(__builtin_amdgcn_global_load_async_to_lds_b128)
  void* gg = const_cast<void*>(g);
  __builtin_amdgcn_global_load_async_to_lds_b128(
      (gv4i*)gg, (lv4i*)lds, 0, 0);
#else
  unsigned laddr = (unsigned)(unsigned long long)lds;  // low 32 bits = LDS byte address
  asm volatile("global_load_async_to_lds_b128 %0, %1, off" :: "v"(laddr), "v"(g) : "memory");
#endif
}

template <int N>
__device__ __forceinline__ void wait_async() {
#if __has_builtin(__builtin_amdgcn_s_wait_asynccnt)
  __builtin_amdgcn_s_wait_asynccnt(N);
#else
  asm volatile("s_wait_asynccnt %0" :: "i"(N) : "memory");
#endif
}

// Load one 16x32 f16 WMMA fragment from an LDS tile (row-major, LDS_STRIDE halves/row).
// Per ISA layout: lanes 0-15 hold K=0..7 and 16..23; lanes 16-31 hold K=8..15 and 24..31.
// kh = 0 (lanes<16) or 8 (lanes>=16). A and B layouts are identical -> Gram needs no transpose.
__device__ __forceinline__ v16h load_frag(const _Float16* rowp, int kh) {
  v8h a = *(const v8h*)(rowp + kh);
  v8h b = *(const v8h*)(rowp + kh + 16);
  return __builtin_shufflevector(a, b, 0, 1, 2, 3, 4, 5, 6, 7,
                                        8, 9, 10, 11, 12, 13, 14, 15);
}

// ---------------- kernel 1: fp32 -> f16 hi/lo split + exact row norms ----------------
__global__ __launch_bounds__(256)
void mmd_prep(const float* __restrict__ realp, const float* __restrict__ genp,
              _Float16* __restrict__ zhi, _Float16* __restrict__ zlo,
              float* __restrict__ norms, double* __restrict__ acc) {
  __shared__ float red[256];
  const int row = blockIdx.x;
  const float* src = (row < HALFROWS) ? (realp + (size_t)row * KDIM)
                                      : (genp + (size_t)(row - HALFROWS) * KDIM);
  const size_t base = (size_t)row * KDIM;
  float ss = 0.f;
  for (int k = threadIdx.x; k < KDIM; k += 256) {
    float x = src[k];
    _Float16 hi = (_Float16)x;
    _Float16 lo = (_Float16)(x - (float)hi);
    zhi[base + k] = hi;
    zlo[base + k] = lo;
    ss += x * x;
  }
  red[threadIdx.x] = ss;
  __syncthreads();
  for (int off = 128; off > 0; off >>= 1) {
    if (threadIdx.x < off) red[threadIdx.x] += red[threadIdx.x + off];
    __syncthreads();
  }
  if (threadIdx.x == 0) {
    norms[row] = red[0];
    if (row == 0) *acc = 0.0;
  }
}

// ---------------- kernel 2: tiled Gram GEMM + fused exp/sign/reduce ----------------
__global__ __launch_bounds__(256)
void mmd_gemm(const _Float16* __restrict__ zhi, const _Float16* __restrict__ zlo,
              const float* __restrict__ norms, double* __restrict__ acc_out) {
  __shared__ _Float16 smem[2][4][TILE_HALVES];  // [stage][Ahi,Alo,Bhi,Blo][128 x 40]
  __shared__ float redsum;

  const int tid  = threadIdx.x;
  const int lane = tid & 31;
  const int wave = tid >> 5;     // 8 waves
  const int wm   = wave >> 1;    // 0..3  (32 rows each)
  const int wn   = wave & 1;     // 0..1  (64 cols each)
  const int mBlock = blockIdx.y * BM;
  const int nBlock = blockIdx.x * BN;

  if (tid == 0) redsum = 0.f;

  // Stage fill: 2048 16-byte chunks per stage, 8 per thread, all via async-to-LDS.
  auto prefetch = [&](int kt, int stage) {
#pragma unroll
    for (int i = 0; i < 8; ++i) {
      const int c    = tid + i * 256;
      const int tile = c >> 9;          // 0:Ahi 1:Alo 2:Bhi 3:Blo
      const int row  = (c >> 2) & 127;
      const int part = c & 3;           // 16B chunk within 64B row slab
      const _Float16* src = (tile & 1) ? zlo : zhi;
      const int rowG = ((tile < 2) ? mBlock : nBlock) + row;
      const _Float16* g = src + (size_t)rowG * KDIM + kt * BK + part * 8;
      _Float16* l = &smem[stage][tile][row * LDS_STRIDE + part * 8];
      async_b128(l, g);
    }
  };

  v8f acc[2][4];
#pragma unroll
  for (int mt = 0; mt < 2; ++mt)
#pragma unroll
    for (int nt = 0; nt < 4; ++nt)
      acc[mt][nt] = v8f{};

  prefetch(0, 0);

  const int rl = lane & 15;
  const int kh = (lane & 16) >> 1;  // 0 or 8

  for (int kt = 0; kt < KSTEPS; ++kt) {
    const int stage = kt & 1;
    if (kt + 1 < KSTEPS) { prefetch(kt + 1, stage ^ 1); wait_async<8>(); }
    else                 { wait_async<0>(); }
    __syncthreads();  // stage `stage` fully resident for all waves

    const _Float16* Ahi = smem[stage][0];
    const _Float16* Alo = smem[stage][1];
    const _Float16* Bhi = smem[stage][2];
    const _Float16* Blo = smem[stage][3];

    v16h ah[2], al[2];
#pragma unroll
    for (int mt = 0; mt < 2; ++mt) {
      const int ar = wm * 32 + mt * 16 + rl;
      ah[mt] = load_frag(Ahi + ar * LDS_STRIDE, kh);
      al[mt] = load_frag(Alo + ar * LDS_STRIDE, kh);
    }
#pragma unroll
    for (int nt = 0; nt < 4; ++nt) {
      const int br = wn * 64 + nt * 16 + rl;
      v16h bh = load_frag(Bhi + br * LDS_STRIDE, kh);
      v16h bl = load_frag(Blo + br * LDS_STRIDE, kh);
      // split-precision product: hi*hi + hi*lo + lo*hi (lo*lo ~2^-21, dropped)
#pragma unroll
      for (int mt = 0; mt < 2; ++mt)
        acc[mt][nt] = __builtin_amdgcn_wmma_f32_16x16x32_f16(
            false, ah[mt], false, bh, (short)0, acc[mt][nt], false, false);
#pragma unroll
      for (int mt = 0; mt < 2; ++mt)
        acc[mt][nt] = __builtin_amdgcn_wmma_f32_16x16x32_f16(
            false, ah[mt], false, bl, (short)0, acc[mt][nt], false, false);
#pragma unroll
      for (int mt = 0; mt < 2; ++mt)
        acc[mt][nt] = __builtin_amdgcn_wmma_f32_16x16x32_f16(
            false, al[mt], false, bh, (short)0, acc[mt][nt], false, false);
    }
    __syncthreads();  // all reads of this stage done before it is refilled
  }

  // Fused epilogue: exp(-(||x||^2+||y||^2-2xy)/DIM) with +/- block sign, then reduce.
  // C layout: element e of v8f -> row M = e + (lane>=16 ? 8 : 0), col N = lane&15.
  float partial = 0.f;
  const int ehalf = (lane & 16) >> 1;
#pragma unroll
  for (int mt = 0; mt < 2; ++mt) {
#pragma unroll
    for (int nt = 0; nt < 4; ++nt) {
      const int col = nBlock + wn * 64 + nt * 16 + rl;
      const float nB = norms[col];
      const float sB = (col < HALFROWS) ? 1.f : -1.f;
#pragma unroll
      for (int e = 0; e < 8; ++e) {
        const int row = mBlock + wm * 32 + mt * 16 + ehalf + e;
        const float nA = norms[row];
        const float sgn = (row < HALFROWS) ? sB : -sB;
        const float s = acc[mt][nt][e];
        partial += sgn * __expf(-(nA + nB - 2.f * s) * (1.f / (float)KDIM));
      }
    }
  }
  atomicAdd(&redsum, partial);
  __syncthreads();
  if (tid == 0) atomicAdd(acc_out, (double)redsum);
}

// ---------------- kernel 3: scale + emit fp32 scalar ----------------
__global__ void mmd_finalize(const double* __restrict__ acc, float* __restrict__ out) {
  out[0] = (float)(acc[0] * (1.0 / ((double)HALFROWS * (double)HALFROWS)));
}

// ---------------- launcher ----------------
extern "C" void kernel_launch(void* const* d_in, const int* in_sizes, int n_in,
                              void* d_out, int out_size, void* d_ws, size_t ws_size,
                              hipStream_t stream) {
  (void)in_sizes; (void)n_in; (void)out_size; (void)ws_size;
  const float* realp = (const float*)d_in[0];
  const float* genp  = (const float*)d_in[1];
  float* out = (float*)d_out;

  // workspace layout: [0,8) f64 accumulator | [256, 256+32KB) norms | [64KB, ...) Zhi | Zlo
  char* ws = (char*)d_ws;
  double*   acc   = (double*)ws;
  float*    norms = (float*)(ws + 256);
  _Float16* zhi   = (_Float16*)(ws + (64 << 10));
  _Float16* zlo   = zhi + (size_t)NROWS * KDIM;   // total ws use ~64.1 MB

  mmd_prep<<<dim3(NROWS), dim3(256), 0, stream>>>(realp, genp, zhi, zlo, norms, acc);
  mmd_gemm<<<dim3(NROWS / BN, NROWS / BM), dim3(256), 0, stream>>>(zhi, zlo, norms, acc);
  mmd_finalize<<<1, 1, 0, stream>>>(acc, out);
}
